// Model_13082470384329
// MI455X (gfx1250) — compile-verified, matrix operand order checked by reading, not run on hardware
//
#include <hip/hip_runtime.h>
#include <math.h>

// ---------------------------------------------------------------------------
// Types for CDNA5 WMMA (wave32, 16x16x32 bf16 -> f32)
// ---------------------------------------------------------------------------
typedef __attribute__((ext_vector_type(16))) __bf16 v16bf;
typedef __attribute__((ext_vector_type(8)))  float  v8f;

union Frag {
    v16bf v;
    unsigned int u[8];
};

static __device__ __forceinline__ unsigned short f2bf(float x) {
    // round-to-nearest-even f32 -> bf16
    unsigned int u = __float_as_uint(x);
    u += 0x7FFFu + ((u >> 16) & 1u);
    return (unsigned short)(u >> 16);
}

// ---------------------------------------------------------------------------
// Zero-fill (grid-stride)
// ---------------------------------------------------------------------------
__global__ void zero_f32(float* __restrict__ p, long long n) {
    long long stride = (long long)gridDim.x * blockDim.x;
    for (long long i = (long long)blockIdx.x * blockDim.x + threadIdx.x; i < n; i += stride)
        p[i] = 0.0f;
}

// ---------------------------------------------------------------------------
// COO SPMM scatter:  acc[row[e], :] += val[e] * X[col[e], :]
// One block per edge; blockDim = F/4; each thread handles 4 features.
// row/col/val loads are blockIdx-uniform -> scalarized by the compiler.
// ---------------------------------------------------------------------------
__global__ void spmm_edge(const int* __restrict__ row, const int* __restrict__ col,
                          const float* __restrict__ val, const float* __restrict__ X,
                          float* __restrict__ acc, int F) {
    const int e = blockIdx.x;
    const int f = threadIdx.x * 4;
    const float v = val[e];
    const long long c = col[e];
    const long long r = row[e];
    const float4 x = *(const float4*)(X + c * F + f);
    float* dst = acc + r * F + f;
    atomicAdd(dst + 0, v * x.x);
    atomicAdd(dst + 1, v * x.y);
    atomicAdd(dst + 2, v * x.z);
    atomicAdd(dst + 3, v * x.w);
}

// ---------------------------------------------------------------------------
// WMMA GEMM with optional row gather, bias and activation.
//   C[m, n] = act( sum_k A[rowsel(m), k] * W[k, n] + bias[n] )
// Block computes a 64x64 output tile; 8 waves, each wave owns 16(M) x 32(N)
// via two 16x16x32 bf16 WMMA accumulators. K must be a multiple of 32,
// Nout a multiple of 64.
// ---------------------------------------------------------------------------
#define ACT_NONE    0
#define ACT_LRELU   1
#define ACT_SIGMOID 2

__global__ __launch_bounds__(256)
void gemm_bias_act(const float* __restrict__ A, int lda,
                   const int* __restrict__ gather,   // nullptr -> identity rows
                   const float* __restrict__ W,      // [K, Nout] row-major
                   const float* __restrict__ bias,   // [Nout]
                   float* __restrict__ C, int ldc,
                   int M, int K, int Nout, int act, float slope) {
    __shared__ unsigned short lA[64][34];   // [m][k] bf16, padded
    __shared__ unsigned short lB[64][34];   // [n][k] bf16 (W tile pre-transposed)

    const int t    = threadIdx.x;
    const int m0   = blockIdx.y * 64;
    const int n0   = blockIdx.x * 64;
    const int wave = t >> 5;
    const int lane = t & 31;
    const int half = lane >> 4;
    const int l16  = lane & 15;
    const int wm   = (wave & 3) * 16;   // wave M offset inside block tile
    const int wn   = (wave >> 2) * 32;  // wave N offset inside block tile

    // A staging: thread -> (row ar, 8 consecutive k at ak)
    const int ar = t >> 2;
    const int ak = (t & 3) * 8;
    int arow = m0 + ar;
    if (arow >= M) arow = M - 1;                       // clamp (writes masked later)
    const long long aSrc = gather ? (long long)gather[arow] : (long long)arow;
    const float* Aptr = A + aSrc * (long long)lda;

    // B staging: thread -> (k row bk, 8 consecutive n at bn); stored transposed
    const int bk = t >> 3;
    const int bn = (t & 7) * 8;

    v8f c0 = {};
    v8f c1 = {};

    for (int k0 = 0; k0 < K; k0 += 32) {
        // ---- stage A tile (64x32) as bf16 ----
        float4 a0 = *(const float4*)(Aptr + k0 + ak);
        float4 a1 = *(const float4*)(Aptr + k0 + ak + 4);
        lA[ar][ak + 0] = f2bf(a0.x); lA[ar][ak + 1] = f2bf(a0.y);
        lA[ar][ak + 2] = f2bf(a0.z); lA[ar][ak + 3] = f2bf(a0.w);
        lA[ar][ak + 4] = f2bf(a1.x); lA[ar][ak + 5] = f2bf(a1.y);
        lA[ar][ak + 6] = f2bf(a1.z); lA[ar][ak + 7] = f2bf(a1.w);
        // ---- stage W tile (32x64) as bf16, transposed to [n][k] ----
        const float* Wp = W + (long long)(k0 + bk) * Nout + n0 + bn;
        float4 b0v = *(const float4*)(Wp);
        float4 b1v = *(const float4*)(Wp + 4);
        lB[bn + 0][bk] = f2bf(b0v.x); lB[bn + 1][bk] = f2bf(b0v.y);
        lB[bn + 2][bk] = f2bf(b0v.z); lB[bn + 3][bk] = f2bf(b0v.w);
        lB[bn + 4][bk] = f2bf(b1v.x); lB[bn + 5][bk] = f2bf(b1v.y);
        lB[bn + 6][bk] = f2bf(b1v.z); lB[bn + 7][bk] = f2bf(b1v.w);
        __syncthreads();

        // ---- build fragments per ISA 16-bit A/B layout ----
        // pair p holds bf16 K-values {ks, ks+1}:
        //   p<4 : ks = 2p + 8*half       (K 0..7 lanes0-15 / K 8..15 lanes16-31)
        //   p>=4: ks = 2p + 8 + 8*half   (K 16..31 split by half)
        Frag fa, fb0, fb1;
#pragma unroll
        for (int p = 0; p < 8; ++p) {
            const int ks = (p < 4) ? (2 * p + half * 8) : (2 * p + 8 + half * 8);
            fa.u[p]  = *(const unsigned int*)&lA[wm + l16][ks];
            fb0.u[p] = *(const unsigned int*)&lB[wn + l16][ks];
            fb1.u[p] = *(const unsigned int*)&lB[wn + 16 + l16][ks];
        }
        c0 = __builtin_amdgcn_wmma_f32_16x16x32_bf16(false, fa.v, false, fb0.v,
                                                     (short)0, c0, false, false);
        c1 = __builtin_amdgcn_wmma_f32_16x16x32_bf16(false, fa.v, false, fb1.v,
                                                     (short)0, c1, false, false);
        __syncthreads();
    }

    // ---- epilogue: bias + activation, D layout: vgpr r -> m = r + 8*half, n = l16
#pragma unroll
    for (int r = 0; r < 8; ++r) {
        const int m   = wm + r + 8 * half;
        const int row = m0 + m;
        if (row < M) {
            const int n = n0 + wn + l16;
            float x0 = c0[r] + bias[n];
            float x1 = c1[r] + bias[n + 16];
            if (act == ACT_LRELU) {
                x0 = (x0 > 0.0f) ? x0 : slope * x0;
                x1 = (x1 > 0.0f) ? x1 : slope * x1;
            } else if (act == ACT_SIGMOID) {
                x0 = 1.0f / (1.0f + expf(-x0));
                x1 = 1.0f / (1.0f + expf(-x1));
            }
            float* cp = C + (long long)row * ldc + n;
            cp[0]  = x0;
            cp[16] = x1;
        }
    }
}

// ---------------------------------------------------------------------------
// Gated blend: h = g*nxt + (1-g)*h   (g already sigmoid'ed)
// ---------------------------------------------------------------------------
__global__ void gate_combine(float* __restrict__ h, const float* __restrict__ nxt,
                             const float* __restrict__ g, long long n) {
    long long stride = (long long)gridDim.x * blockDim.x;
    for (long long i = (long long)blockIdx.x * blockDim.x + threadIdx.x; i < n; i += stride) {
        const float gv = g[i];
        h[i] = gv * nxt[i] + (1.0f - gv) * h[i];
    }
}

// ---------------------------------------------------------------------------
// Row gather: out[b, 0:Dd] = src[ids[b], 0:Dd]
// ---------------------------------------------------------------------------
__global__ void gather_rows(float* __restrict__ out, int ldo,
                            const float* __restrict__ src,
                            const int* __restrict__ ids, int B, int Dd) {
    long long n = (long long)B * Dd;
    long long stride = (long long)gridDim.x * blockDim.x;
    for (long long i = (long long)blockIdx.x * blockDim.x + threadIdx.x; i < n; i += stride) {
        const int b = (int)(i / Dd);
        const int f = (int)(i % Dd);
        out[(long long)b * ldo + f] = src[(long long)ids[b] * Dd + f];
    }
}

// ---------------------------------------------------------------------------
// Launch
// ---------------------------------------------------------------------------
extern "C" void kernel_launch(void* const* d_in, const int* in_sizes, int n_in,
                              void* d_out, int out_size, void* d_ws, size_t ws_size,
                              hipStream_t stream) {
    const float* ent_lit  = (const float*)d_in[0];
    const float* rel_lit  = (const float*)d_in[1];
    const float* attr_lit = (const float*)d_in[2];
    const float* ent_val  = (const float*)d_in[3];
    const float* rel_val  = (const float*)d_in[4];
    const float* attr_val = (const float*)d_in[5];
    const float* W_ent    = (const float*)d_in[6];
    const float* b_ent    = (const float*)d_in[7];
    const float* W_rel    = (const float*)d_in[8];
    const float* b_rel    = (const float*)d_in[9];
    const float* W_attr   = (const float*)d_in[10];
    const float* b_attr   = (const float*)d_in[11];
    const float* W_x      = (const float*)d_in[12];
    const float* b_x      = (const float*)d_in[13];
    const float* W_lit    = (const float*)d_in[14];
    const float* b_lit    = (const float*)d_in[15];
    const float* W_gate   = (const float*)d_in[16];
    const float* b_gate   = (const float*)d_in[17];
    const int* ent_row    = (const int*)d_in[18];
    const int* ent_col    = (const int*)d_in[19];
    const int* rel_row    = (const int*)d_in[20];
    const int* rel_col    = (const int*)d_in[21];
    const int* attr_row   = (const int*)d_in[22];
    const int* attr_col   = (const int*)d_in[23];
    const int* ent_ids    = (const int*)d_in[24];

    const int E = in_sizes[3];
    const int D = in_sizes[7];
    const int F = in_sizes[6] / D;
    const int N = in_sizes[0] / F;
    const int B = in_sizes[24];
    const int ldo = 4 * D;
    const float slope = 0.01f;

    float* out = (float*)d_out;

    // Workspace layout (floats):
    //   accF : N*F   (big SPMM accumulator; later aliased by nxt/gbuf)
    //   h    : N*D
    //   accD : N*D
    float* accF = (float*)d_ws;
    float* h    = accF + (size_t)N * F;
    float* accD = h + (size_t)N * D;
    float* nxt  = accF;                       // alias (accF free after ent GEMM)
    float* gbuf = accF + (size_t)N * D;       // alias

    const long long nF = (long long)N * F;
    const long long nD = (long long)N * D;
    const int ZB = 4096;

    dim3 gemmGridB(D / 64, (B + 63) / 64);    // gathered-row GEMMs
    dim3 gemmGridN(D / 64, (N + 63) / 64);    // full-node GEMMs

    // ---- rel path: spmm -> gathered GEMM into out[:, 2D:3D)
    zero_f32<<<ZB, 256, 0, stream>>>(accF, nF);
    spmm_edge<<<E, F / 4, 0, stream>>>(rel_row, rel_col, rel_val, rel_lit, accF, F);
    gemm_bias_act<<<gemmGridB, 256, 0, stream>>>(accF, F, ent_ids, W_rel, b_rel,
                                                 out + 2 * D, ldo, B, F, D, ACT_LRELU, slope);

    // ---- attr path: spmm -> gathered GEMM into out[:, 3D:4D)
    zero_f32<<<ZB, 256, 0, stream>>>(accF, nF);
    spmm_edge<<<E, F / 4, 0, stream>>>(attr_row, attr_col, attr_val, attr_lit, accF, F);
    gemm_bias_act<<<gemmGridB, 256, 0, stream>>>(accF, F, ent_ids, W_attr, b_attr,
                                                 out + 3 * D, ldo, B, F, D, ACT_LRELU, slope);

    // ---- ent_literals: gathered GEMM (no activation) into out[:, 0:D)
    gemm_bias_act<<<gemmGridB, 256, 0, stream>>>(ent_lit, F, ent_ids, W_lit, b_lit,
                                                 out, ldo, B, F, D, ACT_NONE, slope);

    // ---- ent hop 0: spmm -> full GEMM -> h[N,D]
    zero_f32<<<ZB, 256, 0, stream>>>(accF, nF);
    spmm_edge<<<E, F / 4, 0, stream>>>(ent_row, ent_col, ent_val, ent_lit, accF, F);
    gemm_bias_act<<<gemmGridN, 256, 0, stream>>>(accF, F, nullptr, W_ent, b_ent,
                                                 h, D, N, F, D, ACT_LRELU, slope);

    // ---- hops 1..2: gated aggregation
    for (int hop = 1; hop < 3; ++hop) {
        zero_f32<<<ZB, 256, 0, stream>>>(accD, nD);
        spmm_edge<<<E, D / 4, 0, stream>>>(ent_row, ent_col, ent_val, h, accD, D);
        gemm_bias_act<<<gemmGridN, 256, 0, stream>>>(accD, D, nullptr, W_x, b_x,
                                                     nxt, D, N, D, D, ACT_LRELU, slope);
        gemm_bias_act<<<gemmGridN, 256, 0, stream>>>(h, D, nullptr, W_gate, b_gate,
                                                     gbuf, D, N, D, D, ACT_SIGMOID, slope);
        gate_combine<<<ZB, 256, 0, stream>>>(h, nxt, gbuf, nD);
    }

    // ---- ent_n_hop gather into out[:, D:2D)
    gather_rows<<<ZB, 256, 0, stream>>>(out + D, ldo, h, ent_ids, B, D);
}
